// PK1Block_12120397709902
// MI455X (gfx1250) — matrix operand order repaired and checked by reading, not compiled, not run on hardware
//
#include <hip/hip_runtime.h>
#include <hip/hip_bf16.h>

// ---------------- problem constants ----------------
constexpr int kB    = 2;
constexpr int kS    = 2048;
constexpr int kDim  = 2048;
constexpr int kNH   = 16;
constexpr int kNKV  = 4;
constexpr int kHD   = 128;
constexpr int kNE   = 8;
constexpr int kTopK = 2;
constexpr int kHid  = 1024;
constexpr int kShid = 1024;
constexpr int kT    = kB * kS;        // 4096 tokens

// ---------------- WMMA plumbing ----------------
typedef __attribute__((ext_vector_type(16))) _Float16 v16h;
typedef __attribute__((ext_vector_type(8)))  float    v8f;

union Frag16 { v16h v; uint4 u[2]; };

__device__ __forceinline__ v8f wmma32f16(v16h a, v16h b, v8f c) {
  return __builtin_amdgcn_wmma_f32_16x16x32_f16(
      /*neg_a=*/false, a, /*neg_b=*/false, b,
      /*c_mod=*/(short)0, c, /*reuse_a=*/false, /*reuse_b=*/false);
}

// gfx1250 async global->LDS copy (ASYNCcnt); 16 bytes per lane.
__device__ __forceinline__ void async_ld_b128(unsigned lds_off, const void* gaddr) {
  asm volatile("global_load_async_to_lds_b128 %0, %1, off"
               :: "v"(lds_off), "v"((unsigned long long)(uintptr_t)gaddr)
               : "memory");
}
__device__ __forceinline__ void wait_async0() {
  asm volatile("s_wait_asynccnt 0" ::: "memory");
}

#define LSTR  48   // LDS stride (halves) for attention tiles: 96B rows
#define LSTRB 72   // LDS stride (halves) for BK=64 tiles: 144B rows (16B-aligned)

// ======================================================================
// Dense GEMM: C[M,N] = A[M,K](f16) * B[K,N](f32 weights -> f16)
// Stage: BM=128, BN=64, BK=64. 8 waves, each 16 rows x 64 cols
// -> 8 WMMAs per stage per wave, A tile moved via async-to-LDS.
// MODE 0: store f16.  MODE 1: f32 = res + C.  MODE 2: f16 = silu(mul)*C.
// MODE 3: atomicAdd into f32.
// M mult of 128; N mult of 64; K mult of 64.
// ======================================================================
template<int MODE>
__global__ __launch_bounds__(256)
void gemm_f16(const _Float16* __restrict__ A, const float* __restrict__ Bw,
              int M, int N, int K,
              _Float16* __restrict__ outH, float* __restrict__ outF,
              const float* __restrict__ resF, const _Float16* __restrict__ mulH)
{
  __shared__ _Float16 As[128 * LSTRB];   // [row][k]  18 KB
  __shared__ _Float16 Bs[64 * LSTRB];    // [n][k]     9 KB
  const int tid  = threadIdx.x;
  const int lane = tid & 31;
  const int wv   = tid >> 5;
  const int half = lane >> 4;
  const int lx   = lane & 15;
  const int m0   = blockIdx.y * 128;
  const int n0   = blockIdx.x * 64;
  const int rowT = wv * 16;              // each wave owns 16 rows, all 64 cols

  const int ar = tid >> 1, ac = (tid & 1) * 32;  // A coop: 32 halves/thread
  const int kp = tid >> 3, np = (tid & 7) * 8;   // B coop: k-pair x 8 n/thread

  v8f c[4];
#pragma unroll
  for (int nt = 0; nt < 4; ++nt) c[nt] = (v8f){};

  for (int k0 = 0; k0 < K; k0 += 64) {
    // ---- A tile: async global->LDS (4 x b128 per lane) ----
    {
      const _Float16* gp = A + (size_t)(m0 + ar) * K + k0 + ac;
      unsigned lo = (unsigned)(uintptr_t)&As[ar * LSTRB + ac];
      async_ld_b128(lo,      gp);
      async_ld_b128(lo + 16, gp + 8);
      async_ld_b128(lo + 32, gp + 16);
      async_ld_b128(lo + 48, gp + 24);
    }
    // ---- B tile: fp32 -> f16, packed pair stores (k, k+1) ----
    {
      const float* r0 = Bw + (size_t)(k0 + 2 * kp) * N + n0 + np;
      const float* r1 = r0 + N;
      float4 a0 = ((const float4*)r0)[0], a1 = ((const float4*)r0)[1];
      float4 b0 = ((const float4*)r1)[0], b1 = ((const float4*)r1)[1];
      float f0[8] = {a0.x, a0.y, a0.z, a0.w, a1.x, a1.y, a1.z, a1.w};
      float f1[8] = {b0.x, b0.y, b0.z, b0.w, b1.x, b1.y, b1.z, b1.w};
#pragma unroll
      for (int i = 0; i < 8; ++i) {
        union { unsigned u; _Float16 h[2]; } pk;
        pk.h[0] = (_Float16)f0[i];
        pk.h[1] = (_Float16)f1[i];
        *(unsigned*)&Bs[(np + i) * LSTRB + 2 * kp] = pk.u;
      }
    }
    if (k0 + 64 < K)   // gfx1250 global_prefetch_b8 for the next B tile
      __builtin_prefetch(Bw + (size_t)(k0 + 64 + 2 * kp) * N + n0 + np, 0, 0);

    wait_async0();
    __syncthreads();

#pragma unroll
    for (int ks = 0; ks < 64; ks += 32) {
      Frag16 a;
      a.u[0] = *(const uint4*)&As[(rowT + lx) * LSTRB + ks + half * 8];
      a.u[1] = *(const uint4*)&As[(rowT + lx) * LSTRB + ks + 16 + half * 8];
#pragma unroll
      for (int nt = 0; nt < 4; ++nt) {
        Frag16 b;
        b.u[0] = *(const uint4*)&Bs[(nt * 16 + lx) * LSTRB + ks + half * 8];
        b.u[1] = *(const uint4*)&Bs[(nt * 16 + lx) * LSTRB + ks + 16 + half * 8];
        c[nt] = wmma32f16(a.v, b.v, c[nt]);
      }
    }
    __syncthreads();
  }

#pragma unroll
  for (int i = 0; i < 8; ++i) {
    int m = m0 + rowT + half * 8 + i;
#pragma unroll
    for (int nt = 0; nt < 4; ++nt) {
      size_t ia = (size_t)m * N + n0 + nt * 16 + lx;
      float va = ((const float*)&c[nt])[i];
      if (MODE == 0) { outH[ia] = (_Float16)va; }
      else if (MODE == 1) { outF[ia] = resF[ia] + va; }
      else if (MODE == 2) {
        float ha = (float)mulH[ia];
        float sa = ha / (1.f + __expf(-ha));
        outH[ia] = (_Float16)(sa * va);
      } else {
        atomicAdd(&outF[ia], va);
      }
    }
  }
}

// ======================================================================
// MoE grouped GEMM (up projections): out[(off+m),:kHid] = Xg @ W[e]
// Same 128x64x64 async stage as dense GEMM; A rows gathered per-lane by
// the async copy itself (tail rows clamp to the last valid token; their
// results are discarded in the guarded epilogue).
// doSiluMul: out = silu(mulH) * C  (w3 pass, mulH = h1 buffer)
// ======================================================================
__global__ __launch_bounds__(256)
void moe_gemm_up(const _Float16* __restrict__ Xg, const float* __restrict__ W,
                 const int* __restrict__ cnt, const int* __restrict__ off,
                 const int* __restrict__ etok,
                 _Float16* __restrict__ outH, const _Float16* __restrict__ mulH,
                 int doSiluMul)
{
  __shared__ _Float16 As[128 * LSTRB];
  __shared__ _Float16 Bs[64 * LSTRB];
  const int e = blockIdx.z;
  const int cntE = cnt[e], offE = off[e];
  const int m0 = blockIdx.y * 128;
  if (m0 >= cntE) return;
  const float* Bw = W + (size_t)e * kDim * kHid;
  const int n0 = blockIdx.x * 64;

  const int tid = threadIdx.x, lane = tid & 31, wv = tid >> 5;
  const int half = lane >> 4, lx = lane & 15;
  const int rowT = wv * 16;
  const int ar = tid >> 1, ac = (tid & 1) * 32;
  const int kp = tid >> 3, np = (tid & 7) * 8;

  const int g   = m0 + ar;
  const int tok = etok[offE + ((g < cntE) ? g : (cntE - 1))];  // clamped gather

  v8f c[4];
#pragma unroll
  for (int nt = 0; nt < 4; ++nt) c[nt] = (v8f){};

  for (int k0 = 0; k0 < kDim; k0 += 64) {
    {
      const _Float16* gp = Xg + (size_t)tok * kDim + k0 + ac;
      unsigned lo = (unsigned)(uintptr_t)&As[ar * LSTRB + ac];
      async_ld_b128(lo,      gp);
      async_ld_b128(lo + 16, gp + 8);
      async_ld_b128(lo + 32, gp + 16);
      async_ld_b128(lo + 48, gp + 24);
    }
    {
      const float* r0 = Bw + (size_t)(k0 + 2 * kp) * kHid + n0 + np;
      const float* r1 = r0 + kHid;
      float4 a0 = ((const float4*)r0)[0], a1 = ((const float4*)r0)[1];
      float4 b0 = ((const float4*)r1)[0], b1 = ((const float4*)r1)[1];
      float f0[8] = {a0.x, a0.y, a0.z, a0.w, a1.x, a1.y, a1.z, a1.w};
      float f1[8] = {b0.x, b0.y, b0.z, b0.w, b1.x, b1.y, b1.z, b1.w};
#pragma unroll
      for (int i = 0; i < 8; ++i) {
        union { unsigned u; _Float16 h[2]; } pk;
        pk.h[0] = (_Float16)f0[i];
        pk.h[1] = (_Float16)f1[i];
        *(unsigned*)&Bs[(np + i) * LSTRB + 2 * kp] = pk.u;
      }
    }
    if (k0 + 64 < kDim)
      __builtin_prefetch(Bw + (size_t)(k0 + 64 + 2 * kp) * kHid + n0 + np, 0, 0);

    wait_async0();
    __syncthreads();

#pragma unroll
    for (int ks = 0; ks < 64; ks += 32) {
      Frag16 a;
      a.u[0] = *(const uint4*)&As[(rowT + lx) * LSTRB + ks + half * 8];
      a.u[1] = *(const uint4*)&As[(rowT + lx) * LSTRB + ks + 16 + half * 8];
#pragma unroll
      for (int nt = 0; nt < 4; ++nt) {
        Frag16 b;
        b.u[0] = *(const uint4*)&Bs[(nt * 16 + lx) * LSTRB + ks + half * 8];
        b.u[1] = *(const uint4*)&Bs[(nt * 16 + lx) * LSTRB + ks + 16 + half * 8];
        c[nt] = wmma32f16(a.v, b.v, c[nt]);
      }
    }
    __syncthreads();
  }

#pragma unroll
  for (int i = 0; i < 8; ++i) {
    int gm = m0 + rowT + half * 8 + i;
    if (gm >= cntE) continue;
#pragma unroll
    for (int nt = 0; nt < 4; ++nt) {
      size_t ia = (size_t)(offE + gm) * kHid + n0 + nt * 16 + lx;
      float va = ((const float*)&c[nt])[i];
      if (doSiluMul) {
        float ha = (float)mulH[ia];
        float sa = ha / (1.f + __expf(-ha));
        outH[ia] = (_Float16)(sa * va);
      } else {
        outH[ia] = (_Float16)va;
      }
    }
  }
}

// ======================================================================
// MoE down projection: atomicAdd(outX[tok], gate * (He @ W2[e]))
// ======================================================================
__global__ __launch_bounds__(256)
void moe_gemm_down(const _Float16* __restrict__ He, const float* __restrict__ W2,
                   const int* __restrict__ cnt, const int* __restrict__ off,
                   const int* __restrict__ etok, const float* __restrict__ egw,
                   float* __restrict__ outX)
{
  __shared__ _Float16 As[128 * LSTRB];
  __shared__ _Float16 Bs[64 * LSTRB];
  const int e = blockIdx.z;
  const int cntE = cnt[e], offE = off[e];
  const int m0 = blockIdx.y * 128;
  if (m0 >= cntE) return;
  const float* Bw = W2 + (size_t)e * kHid * kDim;
  const int n0 = blockIdx.x * 64;

  const int tid = threadIdx.x, lane = tid & 31, wv = tid >> 5;
  const int half = lane >> 4, lx = lane & 15;
  const int rowT = wv * 16;
  const int ar = tid >> 1, ac = (tid & 1) * 32;
  const int kp = tid >> 3, np = (tid & 7) * 8;

  const int g = m0 + ar;
  const int rowA = offE + ((g < cntE) ? g : (cntE - 1));   // clamped row

  v8f c[4];
#pragma unroll
  for (int nt = 0; nt < 4; ++nt) c[nt] = (v8f){};

  for (int k0 = 0; k0 < kHid; k0 += 64) {
    {
      const _Float16* gp = He + (size_t)rowA * kHid + k0 + ac;
      unsigned lo = (unsigned)(uintptr_t)&As[ar * LSTRB + ac];
      async_ld_b128(lo,      gp);
      async_ld_b128(lo + 16, gp + 8);
      async_ld_b128(lo + 32, gp + 16);
      async_ld_b128(lo + 48, gp + 24);
    }
    {
      const float* r0 = Bw + (size_t)(k0 + 2 * kp) * kDim + n0 + np;
      const float* r1 = r0 + kDim;
      float4 a0 = ((const float4*)r0)[0], a1 = ((const float4*)r0)[1];
      float4 b0 = ((const float4*)r1)[0], b1 = ((const float4*)r1)[1];
      float f0[8] = {a0.x, a0.y, a0.z, a0.w, a1.x, a1.y, a1.z, a1.w};
      float f1[8] = {b0.x, b0.y, b0.z, b0.w, b1.x, b1.y, b1.z, b1.w};
#pragma unroll
      for (int i = 0; i < 8; ++i) {
        union { unsigned u; _Float16 h[2]; } pk;
        pk.h[0] = (_Float16)f0[i];
        pk.h[1] = (_Float16)f1[i];
        *(unsigned*)&Bs[(np + i) * LSTRB + 2 * kp] = pk.u;
      }
    }
    if (k0 + 64 < kHid)
      __builtin_prefetch(Bw + (size_t)(k0 + 64 + 2 * kp) * kDim + n0 + np, 0, 0);

    wait_async0();
    __syncthreads();

#pragma unroll
    for (int ks = 0; ks < 64; ks += 32) {
      Frag16 a;
      a.u[0] = *(const uint4*)&As[(rowT + lx) * LSTRB + ks + half * 8];
      a.u[1] = *(const uint4*)&As[(rowT + lx) * LSTRB + ks + 16 + half * 8];
#pragma unroll
      for (int nt = 0; nt < 4; ++nt) {
        Frag16 b;
        b.u[0] = *(const uint4*)&Bs[(nt * 16 + lx) * LSTRB + ks + half * 8];
        b.u[1] = *(const uint4*)&Bs[(nt * 16 + lx) * LSTRB + ks + 16 + half * 8];
        c[nt] = wmma32f16(a.v, b.v, c[nt]);
      }
    }
    __syncthreads();
  }

#pragma unroll
  for (int i = 0; i < 8; ++i) {
    int gm = m0 + rowT + half * 8 + i;
    if (gm >= cntE) continue;
    int tok = etok[offE + gm];
    float gv = egw[offE + gm];
#pragma unroll
    for (int nt = 0; nt < 4; ++nt) {
      size_t ia = (size_t)tok * kDim + n0 + nt * 16 + lx;
      atomicAdd(&outX[ia], gv * ((const float*)&c[nt])[i]);
    }
  }
}

// ======================================================================
// Flash attention: one wave handles 16 query rows of one (b,h).
// Q*K^T and P*V via WMMA, online softmax in C-matrix layout.
// ======================================================================
__global__ __launch_bounds__(32)
void flash_attn(const _Float16* __restrict__ qf, const _Float16* __restrict__ kf,
                const _Float16* __restrict__ vf, _Float16* __restrict__ attnO)
{
  __shared__ _Float16 pS[16 * LSTR];      // probs 16x32, row-major (m, kv)
  __shared__ _Float16 vS[128 * LSTR];     // V^T: [hd][kv(32)]
  constexpr int NT = kS / 16;             // 128 q-tiles per (b,h)
  const int blk = blockIdx.x;
  const int qtile = blk % NT;
  const int h = (blk / NT) % kNH;
  const int b = blk / (NT * kNH);
  const int kvh = h / (kNH / kNKV);
  const int lane = threadIdx.x;
  const int half = lane >> 4, lx = lane & 15;

  // Q fragments: 4 hd-chunks of 32
  v16h qfr[4];
  {
    size_t qrow = ((size_t)(b * kS + qtile * 16 + lx) * kNH + h) * kHD;
#pragma unroll
    for (int s = 0; s < 4; ++s) {
      Frag16 a;
      a.u[0] = *(const uint4*)(qf + qrow + s * 32 + half * 8);
      a.u[1] = *(const uint4*)(qf + qrow + s * 32 + 16 + half * 8);
      qfr[s] = a.v;
    }
  }

  v8f o[8];
#pragma unroll
  for (int nt = 0; nt < 8; ++nt) o[nt] = (v8f){};
  float mM[8], lL[8];
#pragma unroll
  for (int i = 0; i < 8; ++i) { mM[i] = -1e30f; lL[i] = 0.f; }

  const float sc = 0.08838834764831845f;  // 1/sqrt(128)
  const int jmax = (qtile * 16 + 15) / 32;

  for (int j = 0; j <= jmax; ++j) {
    // ---- stage V^T tile: lane owns kv = j*32 + lane ----
    {
      const _Float16* vp = vf + ((size_t)(b * kS + j * 32 + lane) * kNKV + kvh) * kHD;
      for (int c8 = 0; c8 < kHD; c8 += 8) {
        union { uint4 u; _Float16 hh[8]; } pk;
        pk.u = *(const uint4*)(vp + c8);
#pragma unroll
        for (int ee = 0; ee < 8; ++ee) vS[(c8 + ee) * LSTR + lane] = pk.hh[ee];
      }
    }
    __syncthreads();

    // ---- scores S = Q * K^T (two 16-wide kv tiles) ----
    v8f s0 = {}; v8f s1 = {};
#pragma unroll
    for (int s = 0; s < 4; ++s) {
      Frag16 kb0, kb1;
      size_t kr0 = ((size_t)(b * kS + j * 32 + lx) * kNKV + kvh) * kHD + s * 32 + half * 8;
      size_t kr1 = ((size_t)(b * kS + j * 32 + 16 + lx) * kNKV + kvh) * kHD + s * 32 + half * 8;
      kb0.u[0] = *(const uint4*)(kf + kr0);
      kb0.u[1] = *(const uint4*)(kf + kr0 + 16);
      kb1.u[0] = *(const uint4*)(kf + kr1);
      kb1.u[1] = *(const uint4*)(kf + kr1 + 16);
      s0 = wmma32f16(qfr[s], kb0.v, s0);
      s1 = wmma32f16(qfr[s], kb1.v, s1);
    }

    // ---- mask + online softmax (row m = half*8+i, col n = lx / lx+16) ----
    const float* sp0 = (const float*)&s0;
    const float* sp1 = (const float*)&s1;
#pragma unroll
    for (int i = 0; i < 8; ++i) {
      int qp = qtile * 16 + half * 8 + i;
      int n0 = j * 32 + lx, n1 = n0 + 16;
      float a0 = (n0 <= qp) ? sp0[i] * sc : -1e30f;
      float a1 = (n1 <= qp) ? sp1[i] * sc : -1e30f;
      float rmx = fmaxf(a0, a1);
      for (int offp = 1; offp < 16; offp <<= 1)
        rmx = fmaxf(rmx, __shfl_xor(rmx, offp, 32));
      float mnew = fmaxf(mM[i], rmx);
      float scl = __expf(mM[i] - mnew);
      float p0v = __expf(a0 - mnew), p1v = __expf(a1 - mnew);
      float rs = p0v + p1v;
      for (int offp = 1; offp < 16; offp <<= 1)
        rs += __shfl_xor(rs, offp, 32);
      lL[i] = lL[i] * scl + rs;
      mM[i] = mnew;
#pragma unroll
      for (int nt = 0; nt < 8; ++nt) ((float*)&o[nt])[i] *= scl;
      pS[(half * 8 + i) * LSTR + lx] = (_Float16)p0v;
      pS[(half * 8 + i) * LSTR + 16 + lx] = (_Float16)p1v;
    }
    __syncthreads();

    // ---- P (16x32) as A fragment, then O += P * V ----
    Frag16 pa;
    pa.u[0] = *(const uint4*)&pS[lx * LSTR + half * 8];
    pa.u[1] = *(const uint4*)&pS[lx * LSTR + 16 + half * 8];
#pragma unroll
    for (int nt = 0; nt < 8; ++nt) {
      Frag16 vb;
      vb.u[0] = *(const uint4*)&vS[(nt * 16 + lx) * LSTR + half * 8];
      vb.u[1] = *(const uint4*)&vS[(nt * 16 + lx) * LSTR + 16 + half * 8];
      o[nt] = wmma32f16(pa.v, vb.v, o[nt]);
    }
    __syncthreads();
  }

  // ---- normalize + store ----
#pragma unroll
  for (int i = 0; i < 8; ++i) {
    float inv = 1.f / lL[i];
    int qp = qtile * 16 + half * 8 + i;
    size_t rb = ((size_t)(b * kS + qp) * kNH + h) * kHD;
#pragma unroll
    for (int nt = 0; nt < 8; ++nt)
      attnO[rb + nt * 16 + lx] = (_Float16)(((const float*)&o[nt])[i] * inv);
  }
}

// ======================================================================
// Elementwise / reduction kernels
// ======================================================================
__global__ __launch_bounds__(256)
void rmsnorm_f16(const float* __restrict__ x, const float* __restrict__ w,
                 _Float16* __restrict__ out)
{
  __shared__ float red[256];
  const int t = blockIdx.x;
  const float* xr = x + (size_t)t * kDim;
  float ss = 0.f;
  for (int d = threadIdx.x; d < kDim; d += 256) { float v = xr[d]; ss += v * v; }
  red[threadIdx.x] = ss; __syncthreads();
  for (int o = 128; o > 0; o >>= 1) {
    if (threadIdx.x < o) red[threadIdx.x] += red[threadIdx.x + o];
    __syncthreads();
  }
  float r = rsqrtf(red[0] / (float)kDim + 1e-6f);
  for (int d = threadIdx.x; d < kDim; d += 256)
    out[(size_t)t * kDim + d] = (_Float16)(xr[d] * r * w[d]);
}

__global__ void rope_q_kernel(_Float16* __restrict__ q, const float* __restrict__ fc,
                              const float* __restrict__ fs)
{
  size_t idx = (size_t)blockIdx.x * blockDim.x + threadIdx.x;
  if (idx >= (size_t)kT * kNH * (kHD / 2)) return;
  int i = (int)(idx % (kHD / 2));
  size_t th = idx / (kHD / 2);
  size_t t = th / kNH;
  int pos = (int)(t % kS);
  float c = fc[(size_t)pos * (kHD / 2) + i], s = fs[(size_t)pos * (kHD / 2) + i];
  _Float16* p = q + th * kHD + 2 * i;
  float t1 = (float)p[0], t2 = (float)p[1];
  p[0] = (_Float16)(t1 * c - t2 * s);
  p[1] = (_Float16)(t1 * s + t2 * c);
}

__global__ void rope_k_kernel(_Float16* __restrict__ k, const float* __restrict__ fc,
                              const float* __restrict__ fs, float* __restrict__ outK)
{
  size_t idx = (size_t)blockIdx.x * blockDim.x + threadIdx.x;
  if (idx >= (size_t)kT * kNKV * (kHD / 2)) return;
  int i = (int)(idx % (kHD / 2));
  size_t th = idx / (kHD / 2);
  size_t t = th / kNKV;
  int pos = (int)(t % kS);
  float c = fc[(size_t)pos * (kHD / 2) + i], s = fs[(size_t)pos * (kHD / 2) + i];
  _Float16* p = k + th * kHD + 2 * i;
  float t1 = (float)p[0], t2 = (float)p[1];
  float r1 = t1 * c - t2 * s, r2 = t1 * s + t2 * c;
  p[0] = (_Float16)r1; p[1] = (_Float16)r2;
  outK[th * kHD + 2 * i] = r1;
  outK[th * kHD + 2 * i + 1] = r2;
}

__global__ void f16_to_f32_kernel(const _Float16* __restrict__ src,
                                  float* __restrict__ dst, size_t n)
{
  size_t i = (size_t)blockIdx.x * blockDim.x + threadIdx.x;
  if (i < n) dst[i] = (float)src[i];
}

__global__ void copy_f32_kernel(const float* __restrict__ src,
                                float* __restrict__ dst, size_t n)
{
  size_t i = (size_t)blockIdx.x * blockDim.x + threadIdx.x;
  if (i < n) dst[i] = src[i];
}

__global__ void zero_ints_kernel(int* a, int n)
{
  int i = blockIdx.x * blockDim.x + threadIdx.x;
  if (i < n) a[i] = 0;
}

__global__ __launch_bounds__(256)
void gate_topk(const _Float16* __restrict__ h2, const float* __restrict__ gw,
               int* __restrict__ topIdx, float* __restrict__ topW, int* __restrict__ cnt)
{
  __shared__ float red[256 * kNE];
  const int t = blockIdx.x;
  float acc[kNE];
#pragma unroll
  for (int e = 0; e < kNE; ++e) acc[e] = 0.f;
  for (int d = threadIdx.x; d < kDim; d += 256) {
    float hv = (float)h2[(size_t)t * kDim + d];
    const float* gr = gw + (size_t)d * kNE;
#pragma unroll
    for (int e = 0; e < kNE; ++e) acc[e] += hv * gr[e];
  }
#pragma unroll
  for (int e = 0; e < kNE; ++e) red[threadIdx.x * kNE + e] = acc[e];
  __syncthreads();
  for (int o = 128; o > 0; o >>= 1) {
    if (threadIdx.x < o)
#pragma unroll
      for (int e = 0; e < kNE; ++e)
        red[threadIdx.x * kNE + e] += red[(threadIdx.x + o) * kNE + e];
    __syncthreads();
  }
  if (threadIdx.x == 0) {
    float lg[kNE];
#pragma unroll
    for (int e = 0; e < kNE; ++e) lg[e] = red[e];
    int i1 = 0;
    for (int e = 1; e < kNE; ++e) if (lg[e] > lg[i1]) i1 = e;
    int i2 = -1;
    for (int e = 0; e < kNE; ++e)
      if (e != i1 && (i2 < 0 || lg[e] > lg[i2])) i2 = e;
    float e2 = __expf(lg[i2] - lg[i1]);
    float inv = 1.f / (1.f + e2);
    topIdx[t * 2] = i1; topIdx[t * 2 + 1] = i2;
    topW[t * 2] = inv;  topW[t * 2 + 1] = e2 * inv;
    atomicAdd(&cnt[i1], 1);
    atomicAdd(&cnt[i2], 1);
  }
}

__global__ void prefix_experts(const int* __restrict__ cnt, int* __restrict__ off,
                               int* __restrict__ pos)
{
  if (threadIdx.x == 0 && blockIdx.x == 0) {
    int a = 0;
    for (int e = 0; e < kNE; ++e) { off[e] = a; a += cnt[e]; pos[e] = 0; }
  }
}

__global__ void scatter_tokens(const int* __restrict__ topIdx, const float* __restrict__ topW,
                               const int* __restrict__ off, int* __restrict__ pos,
                               int* __restrict__ etok, float* __restrict__ egw)
{
  int t = blockIdx.x * blockDim.x + threadIdx.x;
  if (t >= kT) return;
  for (int kk = 0; kk < kTopK; ++kk) {
    int e = topIdx[t * 2 + kk];
    int p = atomicAdd(&pos[e], 1);
    etok[off[e] + p] = t;
    egw[off[e] + p] = topW[t * 2 + kk];
  }
}

// ======================================================================
// Workspace layout
// ======================================================================
constexpr size_t OFF_H    = 0;                                        // T*DIM f16
constexpr size_t OFF_Q    = OFF_H    + (size_t)kT * kDim * 2;         // T*NH*HD f16
constexpr size_t OFF_K    = OFF_Q    + (size_t)kT * kNH * kHD * 2;    // T*NKV*HD f16
constexpr size_t OFF_V    = OFF_K    + (size_t)kT * kNKV * kHD * 2;
constexpr size_t OFF_ATTN = OFF_V    + (size_t)kT * kNKV * kHD * 2;   // T*NH*HD f16
constexpr size_t OFF_X1   = OFF_ATTN + (size_t)kT * kNH * kHD * 2;    // T*DIM f32
constexpr size_t OFF_H2   = OFF_X1   + (size_t)kT * kDim * 4;         // T*DIM f16
constexpr size_t OFF_S1   = OFF_H2   + (size_t)kT * kDim * 2;         // T*SHID f16
constexpr size_t OFF_HE   = OFF_S1   + (size_t)kT * kShid * 2;        // T*TOPK*HID f16
constexpr size_t OFF_TIDX = OFF_HE   + (size_t)kT * kTopK * kHid * 2; // T*2 int
constexpr size_t OFF_TW   = OFF_TIDX + (size_t)kT * 2 * 4;            // T*2 f32
constexpr size_t OFF_CNT  = OFF_TW   + (size_t)kT * 2 * 4;            // NE int
constexpr size_t OFF_OFFE = OFF_CNT  + 64;
constexpr size_t OFF_POS  = OFF_OFFE + 64;
constexpr size_t OFF_ETOK = OFF_POS  + 64;                            // T*2 int
constexpr size_t OFF_EGW  = OFF_ETOK + (size_t)kT * 2 * 4;            // T*2 f32

extern "C" void kernel_launch(void* const* d_in, const int* in_sizes, int n_in,
                              void* d_out, int out_size, void* d_ws, size_t ws_size,
                              hipStream_t stream) {
  (void)in_sizes; (void)n_in; (void)out_size; (void)ws_size;
  const float* x   = (const float*)d_in[0];
  const float* fc  = (const float*)d_in[1];
  const float* fs  = (const float*)d_in[2];
  const float* anw = (const float*)d_in[3];
  const float* fnw = (const float*)d_in[4];
  const float* wq  = (const float*)d_in[5];
  const float* wk  = (const float*)d_in[6];
  const float* wv  = (const float*)d_in[7];
  const float* wo  = (const float*)d_in[8];
  const float* gw  = (const float*)d_in[9];
  const float* w1  = (const float*)d_in[10];
  const float* w2  = (const float*)d_in[11];
  const float* w3  = (const float*)d_in[12];
  const float* sw1 = (const float*)d_in[13];
  const float* sw2 = (const float*)d_in[14];
  const float* sw3 = (const float*)d_in[15];

  float* outX = (float*)d_out;
  float* outK = outX + (size_t)kT * kDim;
  float* outV = outK + (size_t)kT * kNKV * kHD;

  char* ws = (char*)d_ws;
  _Float16* hF    = (_Float16*)(ws + OFF_H);
  _Float16* qF    = (_Float16*)(ws + OFF_Q);
  _Float16* kF    = (_Float16*)(ws + OFF_K);
  _Float16* vF    = (_Float16*)(ws + OFF_V);
  _Float16* attnF = (_Float16*)(ws + OFF_ATTN);
  float*    x1    = (float*)   (ws + OFF_X1);
  _Float16* h2F   = (_Float16*)(ws + OFF_H2);
  _Float16* s1F   = (_Float16*)(ws + OFF_S1);
  _Float16* heF   = (_Float16*)(ws + OFF_HE);
  int*      tIdx  = (int*)     (ws + OFF_TIDX);
  float*    tW    = (float*)   (ws + OFF_TW);
  int*      cnt   = (int*)     (ws + OFF_CNT);
  int*      offE  = (int*)     (ws + OFF_OFFE);
  int*      posE  = (int*)     (ws + OFF_POS);
  int*      etok  = (int*)     (ws + OFF_ETOK);
  float*    egw   = (float*)   (ws + OFF_EGW);

  // 1) attention rmsnorm -> h (f16)
  rmsnorm_f16<<<kT, 256, 0, stream>>>(x, anw, hF);

  // 2) QKV projections (WMMA, async-to-LDS A tiles)
  gemm_f16<0><<<dim3((kNH * kHD) / 64, kT / 128), 256, 0, stream>>>(
      hF, wq, kT, kNH * kHD, kDim, qF, nullptr, nullptr, nullptr);
  gemm_f16<0><<<dim3((kNKV * kHD) / 64, kT / 128), 256, 0, stream>>>(
      hF, wk, kT, kNKV * kHD, kDim, kF, nullptr, nullptr, nullptr);
  gemm_f16<0><<<dim3((kNKV * kHD) / 64, kT / 128), 256, 0, stream>>>(
      hF, wv, kT, kNKV * kHD, kDim, vF, nullptr, nullptr, nullptr);

  // 3) RoPE (q,k in place; k,v f32 outputs)
  {
    size_t nq = (size_t)kT * kNH * (kHD / 2);
    rope_q_kernel<<<(unsigned)((nq + 255) / 256), 256, 0, stream>>>(qF, fc, fs);
    size_t nk = (size_t)kT * kNKV * (kHD / 2);
    rope_k_kernel<<<(unsigned)((nk + 255) / 256), 256, 0, stream>>>(kF, fc, fs, outK);
    size_t nv = (size_t)kT * kNKV * kHD;
    f16_to_f32_kernel<<<(unsigned)((nv + 255) / 256), 256, 0, stream>>>(vF, outV, nv);
  }

  // 4) Flash attention (WMMA, online softmax, GQA)
  flash_attn<<<kB * kNH * (kS / 16), 32, 0, stream>>>(qF, kF, vF, attnF);

  // 5) Output projection + residual: x1 = x + attn @ wo
  gemm_f16<1><<<dim3(kDim / 64, kT / 128), 256, 0, stream>>>(
      attnF, wo, kT, kDim, kNH * kHD, nullptr, x1, x, nullptr);

  // 6) ffn rmsnorm -> h2 (f16)
  rmsnorm_f16<<<kT, 256, 0, stream>>>(x1, fnw, h2F);

  // 7) Router: gate logits, top-2 softmax, expert token lists
  zero_ints_kernel<<<1, 32, 0, stream>>>(cnt, kNE);
  gate_topk<<<kT, 256, 0, stream>>>(h2F, gw, tIdx, tW, cnt);
  prefix_experts<<<1, 1, 0, stream>>>(cnt, offE, posE);
  scatter_tokens<<<(kT + 255) / 256, 256, 0, stream>>>(tIdx, tW, offE, posE, etok, egw);

  // 8) Initialize output x section with x1 (down-projections accumulate into it)
  copy_f32_kernel<<<(unsigned)(((size_t)kT * kDim + 255) / 256), 256, 0, stream>>>(
      x1, outX, (size_t)kT * kDim);

  // 9) MoE expert MLPs (grouped WMMA GEMMs, async gathered A tiles)
  moe_gemm_up<<<dim3(kHid / 64, kT / 128, kNE), 256, 0, stream>>>(
      h2F, w1, cnt, offE, etok, heF, nullptr, 0);                 // h1
  moe_gemm_up<<<dim3(kHid / 64, kT / 128, kNE), 256, 0, stream>>>(
      h2F, w3, cnt, offE, etok, heF, heF, 1);                     // he = silu(h1)*h3
  moe_gemm_down<<<dim3(kDim / 64, kT / 128, kNE), 256, 0, stream>>>(
      heF, w2, cnt, offE, etok, egw, outX);                       // += gate * (he @ w2)

  // 10) Shared FFN
  gemm_f16<0><<<dim3(kShid / 64, kT / 128), 256, 0, stream>>>(
      h2F, sw1, kT, kShid, kDim, s1F, nullptr, nullptr, nullptr);
  gemm_f16<2><<<dim3(kShid / 64, kT / 128), 256, 0, stream>>>(
      h2F, sw3, kT, kShid, kDim, s1F, nullptr, nullptr, s1F);     // s1 = silu(s1)*(h2@sw3)
  gemm_f16<3><<<dim3(kDim / 64, kT / 128), 256, 0, stream>>>(
      s1F, sw2, kT, kDim, kShid, nullptr, outX, nullptr, nullptr); // += s1 @ sw2
}
